// ODS_Layer2_72060961292405
// MI455X (gfx1250) — matrix operand order repaired and checked by reading, not compiled, not run on hardware
//
#include <hip/hip_runtime.h>

typedef __attribute__((ext_vector_type(16))) _Float16 v16h;
typedef __attribute__((ext_vector_type(8)))  _Float16 v8h;
typedef __attribute__((ext_vector_type(8)))  float    v8f;

#define BG 64
#define CH 128
#define HD2 64

__device__ __forceinline__ float leakyf(float v){ return v >= 0.f ? v : 0.2f*v; }
__device__ __forceinline__ float wred_max(float v){ for(int d=16;d>0;d>>=1) v=fmaxf(v,__shfl_xor(v,d,32)); return v; }
__device__ __forceinline__ float wred_sum(float v){ for(int d=16;d>0;d>>=1) v+=__shfl_xor(v,d,32); return v; }

// ---------------------------------------------------------------------------
// Core WMMA GEMM: C[b][m][n] (=|+=) sum_k A[b][m][k] * Bt[b][n][k] (+bias[n])
// One wave per 32x32 output tile (2x2 register blocking -> 4 accumulators,
// 4 v_wmma_f32_16x16x32_f16 per K-step on 2 A-fragments x 2 B-fragments).
// Fragment layouts follow CDNA5 ISA 7.12.2 (A rows per lane with K split by
// lane group; B pre-transposed so lanes read two 8-half K runs (8g, 16+8g); C/D holds
// 8 rows per lane).
// ---------------------------------------------------------------------------
__global__ void k_wmma_gemm(const _Float16* __restrict__ A, long long sAb, int ldA,
                            const _Float16* __restrict__ Bt, long long sBb, int ldB,
                            float* __restrict__ C, long long sCb, int ldC,
                            const float* __restrict__ bias,
                            int K, int accum, int relu)
{
    const int lane = threadIdx.x;
    const int mt = blockIdx.x, nt = blockIdx.y, b = blockIdx.z;
    const int g = lane >> 4, l16 = lane & 15;
    const _Float16* Ab0 = A  + (size_t)b * sAb + (size_t)(mt * 32 + l16) * ldA;
    const _Float16* Ab1 = Ab0 + (size_t)16 * ldA;
    const _Float16* Bb0 = Bt + (size_t)b * sBb + (size_t)(nt * 32 + l16) * ldB;
    const _Float16* Bb1 = Bb0 + (size_t)16 * ldB;
    v8f a00 = {}, a01 = {}, a10 = {}, a11 = {};
    for (int kk = 0; kk < K; kk += 32) {
        union { v16h v; v8h h[2]; } ua0, ua1, ub0, ub1;
        ua0.h[0] = *(const v8h*)(Ab0 + kk + g * 8);
        ua0.h[1] = *(const v8h*)(Ab0 + kk + g * 8 + 16);
        ua1.h[0] = *(const v8h*)(Ab1 + kk + g * 8);
        ua1.h[1] = *(const v8h*)(Ab1 + kk + g * 8 + 16);
        ub0.h[0] = *(const v8h*)(Bb0 + kk + g * 8);
        ub0.h[1] = *(const v8h*)(Bb0 + kk + g * 8 + 16);
        ub1.h[0] = *(const v8h*)(Bb1 + kk + g * 8);
        ub1.h[1] = *(const v8h*)(Bb1 + kk + g * 8 + 16);
        a00 = __builtin_amdgcn_wmma_f32_16x16x32_f16(false, ua0.v, false, ub0.v, (short)0, a00, false, false);
        a01 = __builtin_amdgcn_wmma_f32_16x16x32_f16(false, ua0.v, false, ub1.v, (short)0, a01, false, false);
        a10 = __builtin_amdgcn_wmma_f32_16x16x32_f16(false, ua1.v, false, ub0.v, (short)0, a10, false, false);
        a11 = __builtin_amdgcn_wmma_f32_16x16x32_f16(false, ua1.v, false, ub1.v, (short)0, a11, false, false);
    }
    const int n0 = nt * 32 + l16, n1 = n0 + 16;
    const int m0 = mt * 32 + g * 8;
    const float bv0 = bias ? bias[n0] : 0.0f;
    const float bv1 = bias ? bias[n1] : 0.0f;
    float* Cbase = C + (size_t)b * sCb;
#pragma unroll
    for (int j = 0; j < 8; ++j) {
        float* r0 = Cbase + (size_t)(m0 + j) * ldC;
        float* r1 = Cbase + (size_t)(m0 + 16 + j) * ldC;
        float v00 = a00[j] + bv0, v01 = a01[j] + bv1;
        float v10 = a10[j] + bv0, v11 = a11[j] + bv1;
        if (accum) { v00 += r0[n0]; v01 += r0[n1]; v10 += r1[n0]; v11 += r1[n1]; }
        if (relu)  { v00 = fmaxf(v00, 0.f); v01 = fmaxf(v01, 0.f);
                     v10 = fmaxf(v10, 0.f); v11 = fmaxf(v11, 0.f); }
        r0[n0] = v00; r0[n1] = v01; r1[n0] = v10; r1[n1] = v11;
    }
}

// ------------------------------ utility kernels ----------------------------
__global__ void k_fill(float* p, float v, size_t n){
    size_t t = (size_t)blockIdx.x*256 + threadIdx.x; if (t < n) p[t] = v;
}
__global__ void k_scatter(float* Aadj, const int* ei, int E, int npg){
    int e = blockIdx.x*256 + threadIdx.x; if (e >= E) return;
    int s = ei[e], d = ei[E + e];
    int b = s / npg;
    Aadj[((size_t)b*npg + (d % npg))*(size_t)npg + (s % npg)] = 1.0f;
}
__global__ void k_cvt(_Float16* d, const float* s, size_t n){
    size_t t = (size_t)blockIdx.x*256 + threadIdx.x; if (t < n) d[t] = (_Float16)s[t];
}
// dst[b][c][r] = src[b][r][c]
__global__ void k_trans_cvt(_Float16* d, const float* s, int R, int Cc, int batch, size_t total){
    size_t t = (size_t)blockIdx.x*256 + threadIdx.x; if (t >= total) return;
    size_t per = (size_t)R*Cc; size_t b = t/per; size_t rem = t - b*per;
    int r = (int)(rem / Cc), c = (int)(rem % Cc);
    d[b*per + (size_t)c*R + r] = (_Float16)s[b*per + (size_t)r*Cc + c];
}
// dst[b][i][j] = (transpose? A[b][j][i] : A[b][i][j]); diag forced to 1 if diag
__global__ void k_adj_cvt(_Float16* d, const float* A, int n, int diag, int transpose, size_t total){
    size_t t = (size_t)blockIdx.x*256 + threadIdx.x; if (t >= total) return;
    size_t per = (size_t)n*n; size_t b = t/per; size_t rem = t - b*per;
    int i = (int)(rem / n), j = (int)(rem % n);
    float v = transpose ? A[b*per + (size_t)j*n + i] : A[b*per + (size_t)i*n + j];
    if (diag && i == j) v = 1.0f;
    d[t] = (_Float16)v;
}
__global__ void k_degrees(float* degA, float* degHat, float* dinv, float* invm,
                          const float* A, int n, size_t total){
    size_t t = (size_t)blockIdx.x*256 + threadIdx.x; if (t >= total) return;
    size_t b = t/n; int i = (int)(t % n);
    const float* row = A + (b*(size_t)n + i)*(size_t)n;
    float s = 0.f; for (int j = 0; j < n; ++j) s += row[j];
    float dh = s + 1.0f - row[i];
    degA[t]=s; degHat[t]=dh; dinv[t]=rsqrtf(dh); invm[t]=1.0f/fmaxf(s,1.0f);
}
__global__ void k_rowscale(float* d, const float* s, const float* sc, int n, size_t total){
    size_t t = (size_t)blockIdx.x*256 + threadIdx.x; if (t >= total) return;
    size_t bi = t / CH;
    d[t] = s[t]*sc[bi];
}
__global__ void k_emb_scale_bias(float* emb, int slot, const float* rs, const float* bias,
                                 int n, size_t total){
    size_t t = (size_t)blockIdx.x*256 + threadIdx.x; if (t >= total) return;
    size_t bi = t / CH; int c = (int)(t % CH);
    size_t idx = (bi*6 + slot)*CH + c;
    emb[idx] = emb[idx]*rs[bi] + bias[c];
}
__global__ void k_gin_pre(float* d, const float* x, const float* ax, float eps, size_t total){
    size_t t = (size_t)blockIdx.x*256 + threadIdx.x; if (t >= total) return;
    d[t] = (1.0f + eps)*x[t] + ax[t];
}
__global__ void k_add_bcast(float* d, const float* x, const float* v, int n, size_t total){
    size_t t = (size_t)blockIdx.x*256 + threadIdx.x; if (t >= total) return;
    size_t b = t / ((size_t)n*CH); int c = (int)(t % CH);
    d[t] = x[t] + v[b*CH + c];
}
__global__ void k_vecadd(float* d, const float* a, const float* b, size_t n){
    size_t t = (size_t)blockIdx.x*256 + threadIdx.x; if (t < n) d[t] = a[t] + b[t];
}
__global__ void k_mean_rows(float* out, const float* x, int n, size_t total){
    size_t t = (size_t)blockIdx.x*256 + threadIdx.x; if (t >= total) return;
    size_t b = t / CH; int c = (int)(t % CH);
    const float* p = x + b*(size_t)n*CH + c;
    float s = 0.f; for (int i = 0; i < n; ++i) s += p[(size_t)i*CH];
    out[t] = s/(float)n;
}
__global__ void k_acgate(float* g, const float* ac, const float* w, const float* bb, size_t total){
    size_t t = (size_t)blockIdx.x*256 + threadIdx.x; if (t >= total) return;
    int b = (int)(t/6), o = (int)(t%6);
    float s = bb[o];
    for (int k = 0; k < 5; ++k) s += ac[b*5 + k]*w[k*6 + o];
    g[t] = s;
}
// ------------------------------- GAT kernels -------------------------------
__global__ void k_gat_sd(float* asrcv, float* adstv, const float* hh,
                         const float* wsrc, const float* wdst, int n, size_t total){
    size_t t = (size_t)blockIdx.x*256 + threadIdx.x; if (t >= total) return;
    size_t bi = t/2; int h = (int)(t%2);
    const float* hp = hh + bi*CH + h*HD2;
    float s = 0.f, d = 0.f;
    for (int q = 0; q < HD2; ++q) { s += hp[q]*wsrc[h*HD2+q]; d += hp[q]*wdst[h*HD2+q]; }
    asrcv[t] = s; adstv[t] = d;
}
__global__ void k_gat_alpha(_Float16* alpha, const float* A, const float* asrcv,
                            const float* adstv, int n){
    int i = blockIdx.x, h = blockIdx.y, b = blockIdx.z, lane = threadIdx.x;
    const float* arow = A + ((size_t)b*n + i)*(size_t)n;
    float di = adstv[((size_t)b*n + i)*2 + h];
    float mx = -3.0e38f;
    for (int j = lane; j < n; j += 32){
        bool m = (j==i) || (arow[j] > 0.f);
        if (m){ float e = leakyf(di + asrcv[((size_t)b*n + j)*2 + h]); mx = fmaxf(mx, e); }
    }
    mx = wred_max(mx);
    float sum = 0.f;
    for (int j = lane; j < n; j += 32){
        bool m = (j==i) || (arow[j] > 0.f);
        if (m){ float e = leakyf(di + asrcv[((size_t)b*n + j)*2 + h]); sum += expf(e - mx); }
    }
    sum = wred_sum(sum);
    float inv = 1.0f/sum;
    _Float16* outp = alpha + (((size_t)b*2 + h)*(size_t)n + i)*(size_t)n;
    for (int j = lane; j < n; j += 32){
        bool m = (j==i) || (arow[j] > 0.f);
        float v = 0.f;
        if (m){ float e = leakyf(di + asrcv[((size_t)b*n + j)*2 + h]); v = expf(e - mx)*inv; }
        outp[j] = (_Float16)v;
    }
}
// dst[((b*2+h)*64+d)*n + j] = hh[(b*n+j)*128 + h*64 + d]
__global__ void k_hhT(_Float16* d, const float* hh, int n, size_t total){
    size_t t = (size_t)blockIdx.x*256 + threadIdx.x; if (t >= total) return;
    size_t per = (size_t)2*HD2*n;
    size_t b = t/per; size_t rem = t - b*per;
    int h = (int)(rem/((size_t)HD2*n)); size_t r2 = rem % ((size_t)HD2*n);
    int dd = (int)(r2/n), j = (int)(r2%n);
    d[t] = (_Float16)hh[(b*(size_t)n + j)*CH + h*HD2 + dd];
}
// ----------------------------- attention gate ------------------------------
__global__ void k_scores(float* sc, const float* q, const float* k, const float* g,
                         int n, size_t total){
    size_t t = (size_t)blockIdx.x*256 + threadIdx.x; if (t >= total) return;
    size_t bn = t/36; int ts = (int)(t%36);
    int tt = ts/6, ss = ts%6;
    size_t b = bn/(size_t)n;
    const float* qp = q + (bn*6 + tt)*CH;
    const float* kp = k + (bn*6 + ss)*CH;
    float dot = 0.f;
    for (int c = 0; c < CH; ++c) dot += qp[c]*kp[c];
    sc[t] = dot*0.08838834764831845f*g[b*6 + ss];
}
__global__ void k_attn(float* attOut, const float* sc, size_t total){
    size_t t = (size_t)blockIdx.x*256 + threadIdx.x; if (t >= total) return;
    int tt = (int)(t%6); size_t bn = t/6;
    const float* row = sc + bn*36 + tt*6;
    float a[6];
    for (int s = 0; s < 6; ++s) a[s] = row[s];
    float mx = -3.0e38f; int keep = 0;
    for (int s = 0; s < 6; ++s){
        float as = fabsf(a[s]); int r = 0;
        for (int q = 0; q < 6; ++q){ float aq = fabsf(a[q]); r += (aq > as) || (aq == as && q < s); }
        if (r < 4){ keep |= (1 << s); mx = fmaxf(mx, a[s]); }
    }
    float e[6]; float sum = 0.f;
    for (int s = 0; s < 6; ++s){ e[s] = (keep >> s & 1) ? expf(a[s]-mx) : 0.f; sum += e[s]; }
    float inv = 1.0f/sum;
    float* o = attOut + bn*36 + tt*6;
    for (int s = 0; s < 6; ++s) o[s] = e[s]*inv;
}
__global__ void k_ogate(float* o, const float* att, const float* emb, size_t total){
    size_t t = (size_t)blockIdx.x*256 + threadIdx.x; if (t >= total) return;
    size_t bn = t/CH; int c = (int)(t%CH);
    const float* ar = att + bn*36;
    float ws[6] = {0,0,0,0,0,0};
    for (int tt = 0; tt < 6; ++tt)
        for (int ss = 0; ss < 6; ++ss) ws[ss] += ar[tt*6 + ss];
    float s = 0.f;
    for (int ss = 0; ss < 6; ++ss) s += ws[ss]*emb[(bn*6 + ss)*CH + c];
    o[t] = s;
}
// -------------------------------- ASAP pool --------------------------------
__global__ void k_maskmax(float* xq, const float* x, const float* A, int n){
    int bi = blockIdx.x; int b = bi/n, i = bi%n; int c = threadIdx.x;
    const float* row = A + ((size_t)b*n + i)*(size_t)n;
    float m = -1e30f;
    for (int j = 0; j < n; ++j)
        if (j == i || row[j] > 0.f) m = fmaxf(m, x[((size_t)b*n + j)*CH + c]);
    xq[(size_t)bi*CH + c] = m;
}
__global__ void k_pool_s(_Float16* sbf, const float* scf, const float* A, int n){
    int i = blockIdx.x, b = blockIdx.y, lane = threadIdx.x;
    const float* srow = scf + ((size_t)b*n + i)*(size_t)n;
    const float* arow = A   + ((size_t)b*n + i)*(size_t)n;
    float mx = -3.0e38f;
    for (int j = lane; j < n; j += 32){
        bool m = (j==i) || (arow[j] > 0.f);
        if (m) mx = fmaxf(mx, leakyf(srow[j]));
    }
    mx = wred_max(mx);
    float sum = 0.f;
    for (int j = lane; j < n; j += 32){
        bool m = (j==i) || (arow[j] > 0.f);
        if (m) sum += expf(leakyf(srow[j]) - mx);
    }
    sum = wred_sum(sum);
    float inv = 1.0f/sum;
    _Float16* outp = sbf + ((size_t)b*n + i)*(size_t)n;
    for (int j = lane; j < n; j += 32){
        bool m = (j==i) || (arow[j] > 0.f);
        outp[j] = (_Float16)(m ? expf(leakyf(srow[j]) - mx)*inv : 0.f);
    }
}
__global__ void k_leconv(float* a1, float* b2, float* a3, const float* xn,
                         const float* w1, const float* b1, const float* w2,
                         const float* w3, const float* b3, size_t total){
    size_t t = (size_t)blockIdx.x*256 + threadIdx.x; if (t >= total) return;
    const float* xr = xn + t*CH;
    float s1 = 0.f, s2 = 0.f, s3 = 0.f;
    for (int c = 0; c < CH; ++c){ float v = xr[c]; s1 += v*w1[c]; s2 += v*w2[c]; s3 += v*w3[c]; }
    a1[t] = s1 + b1[0]; b2[t] = s2; a3[t] = s3 + b3[0];
}
__global__ void k_fit(float* fit, const float* A, const float* a1, const float* b2,
                      const float* a3, const float* degHat, int n, size_t total){
    size_t t = (size_t)blockIdx.x*256 + threadIdx.x; if (t >= total) return;
    size_t b = t/n; int i = (int)(t%n);
    const float* row = A + (b*(size_t)n + i)*(size_t)n;
    const float* av = a1 + b*(size_t)n;
    float acc = 0.f;
    for (int j = 0; j < n; ++j){ float w = (j==i) ? 1.0f : row[j]; acc += w*av[j]; }
    float z = acc - degHat[t]*b2[t] + a3[t];
    fit[t] = 1.0f/(1.0f + expf(-z));
}
__global__ void k_topk(int* idx, float* val, const float* fit, int n, int k){
    __shared__ float sf[512];
    int b = blockIdx.x, tid = threadIdx.x;
    sf[tid] = fit[(size_t)b*n + tid];
    __syncthreads();
    float v = sf[tid]; int r = 0;
    for (int j = 0; j < n; ++j){ float o = sf[j]; r += (o > v) || (o == v && j < tid); }
    if (r < k){ idx[b*k + r] = tid; val[b*k + r] = v; }
}
__global__ void k_gather(float* d, const float* xn, const int* idx, const float* val,
                         const float* addv, int n, int k, size_t total){
    size_t t = (size_t)blockIdx.x*256 + threadIdx.x; if (t >= total) return;
    size_t b = t/((size_t)k*CH); int ii = (int)((t/CH)%k); int c = (int)(t%CH);
    int i = idx[b*k + ii];
    float v = xn[(b*(size_t)n + i)*CH + c]*val[b*k + ii];
    if (addv) v += addv[b*CH + c];
    d[t] = v;
}
__global__ void k_A2(float* A2, const float* a2f, const int* idx, int n, int k, size_t total){
    size_t t = (size_t)blockIdx.x*256 + threadIdx.x; if (t >= total) return;
    size_t b = t/((size_t)k*k); int ii = (int)((t/k)%k); int jj = (int)(t%k);
    int i = idx[b*k + ii], j = idx[b*k + jj];
    float v = (ii == jj) ? 0.f : a2f[(b*(size_t)n + i)*(size_t)n + j];
    A2[t] = (v != 0.f) ? 1.0f : 0.0f;
}

// ---------------------------------------------------------------------------
static inline unsigned nblk(size_t n){ return (unsigned)((n + 255)/256); }

// stage param indices (sorted pytree keys)
#define S_ACB 0
#define S_ACW 1
#define S_GAT_ADST 2
#define S_GAT_ASRC 3
#define S_GAT_B 4
#define S_GAT_W 5
#define S_GCN_B 6
#define S_GCN_W 7
#define S_GIN_B1 8
#define S_GIN_B2 9
#define S_GIN_W1 10
#define S_GIN_W2 11
#define S_GR_BREL 12
#define S_GR_WREL 13
#define S_GR_WROOT 14
#define S_KB 15
#define S_KW 16
#define S_LE1_B 17
#define S_LE1_W 18
#define S_LE2_W 19
#define S_LE3_B 20
#define S_LE3_W 21
#define S_LIN_B 22
#define S_LIN_W 23
#define S_PL_B 24
#define S_PL_W 25
#define S_QB 26
#define S_QW 27
#define S_SAGE_BL 28
#define S_SAGE_WL 29
#define S_SAGE_WR 30
// transposed weight indices
enum { W_GCN=0, W_SAGE_WL, W_SAGE_WR, W_GAT, W_GIN1, W_GIN2, W_LIN,
       W_GRREL, W_GRROOT, W_PL, W_KW, W_QW, W_COUNT };

struct WSS {
    hipStream_t st;
    float *A1,*A2,*A4,*scf,*a2f,*emb,*qf,*kf,*scores,*of,*xqf,*xnf,*axf;
    float *tmp1,*tmp2,*hh,*hbuf,*xp,*xs2,*finout;
    float *gm,*v0,*v1,*gmv,*g1,*g2,*degA,*degHat,*dinv,*invm;
    float *asrcv,*adstv,*a1,*b2,*a3,*fit,*val;
    int   *idx;
    _Float16 *AbfRaw,*AbfHat,*AbfHatT,*alphabf,*xbf,*xTbf,*axbf;
    _Float16 *tmpbf,*tmpbf2,*sctbf,*embbf,*hhTbf,*sbf;
};

static void gemm(hipStream_t st, const _Float16* A, long long sAb, int ldA,
                 const _Float16* Bt, long long sBb, int ldB,
                 float* C, long long sCb, int ldC, const float* bias,
                 int M, int N, int K, int batch, int accum, int relu){
    dim3 g((unsigned)(M/32), (unsigned)(N/32), (unsigned)batch);
    k_wmma_gemm<<<g, dim3(32), 0, st>>>(A, sAb, ldA, Bt, sBb, ldB, C, sCb, ldC,
                                        bias, K, accum, relu);
}

static void run_stage(WSS& w, const float* xin, const float* Aadj, int n, float eps,
                      const float* const* P, _Float16* const* WT,
                      const float* gGate, float* attnOut, int kkeep,
                      const float* addv, float* xPooled, float* Aout)
{
    hipStream_t st = w.st;
    const size_t nn = (size_t)BG*n*n, nc = (size_t)BG*n*CH;
    const long long llnn = (long long)n*n, llnc = (long long)n*CH;

    // adjacency variants + degrees
    k_adj_cvt<<<nblk(nn),256,0,st>>>(w.AbfRaw,  Aadj, n, 0, 0, nn);
    k_adj_cvt<<<nblk(nn),256,0,st>>>(w.AbfHat,  Aadj, n, 1, 0, nn);
    k_adj_cvt<<<nblk(nn),256,0,st>>>(w.AbfHatT, Aadj, n, 1, 1, nn);
    k_degrees<<<nblk((size_t)BG*n),256,0,st>>>(w.degA, w.degHat, w.dinv, w.invm, Aadj, n, (size_t)BG*n);
    k_cvt<<<nblk(nc),256,0,st>>>(w.xbf, xin, nc);
    k_trans_cvt<<<nblk(nc),256,0,st>>>(w.xTbf, xin, n, CH, BG, nc);

    // ax = A @ x
    gemm(st, w.AbfRaw, llnn, n, w.xTbf, llnc, n, w.axf, llnc, CH, nullptr, n, CH, n, BG, 0, 0);
    k_cvt<<<nblk(nc),256,0,st>>>(w.axbf, w.axf, nc);

    // GCN (slot 0): dinv_i * (Ahat @ (dinv_j * (x@W))) + b
    gemm(st, w.xbf, 0, CH, WT[W_GCN], 0, CH, w.tmp1, 0, CH, nullptr, BG*n, CH, CH, 1, 0, 0);
    k_rowscale<<<nblk(nc),256,0,st>>>(w.tmp2, w.tmp1, w.dinv, n, nc);
    k_trans_cvt<<<nblk(nc),256,0,st>>>(w.sctbf, w.tmp2, n, CH, BG, nc);
    gemm(st, w.AbfHat, llnn, n, w.sctbf, llnc, n, w.emb + 0*CH, (long long)n*6*CH, 6*CH,
         nullptr, n, CH, n, BG, 0, 0);
    k_emb_scale_bias<<<nblk(nc),256,0,st>>>(w.emb, 0, w.dinv, P[S_GCN_B], n, nc);

    // SAGE (slot 1)
    k_rowscale<<<nblk(nc),256,0,st>>>(w.tmp1, w.axf, w.invm, n, nc);
    k_cvt<<<nblk(nc),256,0,st>>>(w.tmpbf, w.tmp1, nc);
    gemm(st, w.tmpbf, 0, CH, WT[W_SAGE_WL], 0, CH, w.emb + 1*CH, 0, 6*CH,
         P[S_SAGE_BL], BG*n, CH, CH, 1, 0, 0);
    gemm(st, w.xbf, 0, CH, WT[W_SAGE_WR], 0, CH, w.emb + 1*CH, 0, 6*CH,
         nullptr, BG*n, CH, CH, 1, 1, 0);

    // GAT (slot 2)
    gemm(st, w.xbf, 0, CH, WT[W_GAT], 0, CH, w.hh, 0, CH, nullptr, BG*n, CH, CH, 1, 0, 0);
    k_gat_sd<<<nblk((size_t)BG*n*2),256,0,st>>>(w.asrcv, w.adstv, w.hh,
                                                P[S_GAT_ASRC], P[S_GAT_ADST], n, (size_t)BG*n*2);
    k_gat_alpha<<<dim3((unsigned)n,2,BG),32,0,st>>>(w.alphabf, Aadj, w.asrcv, w.adstv, n);
    k_hhT<<<nblk((size_t)BG*2*HD2*n),256,0,st>>>(w.hhTbf, w.hh, n, (size_t)BG*2*HD2*n);
    for (int h = 0; h < 2; ++h){
        gemm(st, w.alphabf + (size_t)h*n*n, (long long)2*n*n, n,
             w.hhTbf + (size_t)h*HD2*n, (long long)2*HD2*n, n,
             w.emb + 2*CH + h*HD2, (long long)n*6*CH, 6*CH,
             P[S_GAT_B] + h*HD2, n, HD2, n, BG, 0, 0);
    }

    // LIN (slot 3)
    gemm(st, w.xbf, 0, CH, WT[W_LIN], 0, CH, w.emb + 3*CH, 0, 6*CH,
         P[S_LIN_B], BG*n, CH, CH, 1, 0, 0);

    // GIN (slot 4)
    k_gin_pre<<<nblk(nc),256,0,st>>>(w.tmp1, xin, w.axf, eps, nc);
    k_cvt<<<nblk(nc),256,0,st>>>(w.tmpbf, w.tmp1, nc);
    gemm(st, w.tmpbf, 0, CH, WT[W_GIN1], 0, CH, w.tmp2, 0, CH, P[S_GIN_B1], BG*n, CH, CH, 1, 0, 1);
    k_cvt<<<nblk(nc),256,0,st>>>(w.tmpbf2, w.tmp2, nc);
    gemm(st, w.tmpbf2, 0, CH, WT[W_GIN2], 0, CH, w.emb + 4*CH, 0, 6*CH,
         P[S_GIN_B2], BG*n, CH, CH, 1, 0, 0);

    // GraphConv (slot 5)
    gemm(st, w.axbf, 0, CH, WT[W_GRREL], 0, CH, w.emb + 5*CH, 0, 6*CH,
         P[S_GR_BREL], BG*n, CH, CH, 1, 0, 0);
    gemm(st, w.xbf, 0, CH, WT[W_GRROOT], 0, CH, w.emb + 5*CH, 0, 6*CH,
         nullptr, BG*n, CH, CH, 1, 1, 0);

    // attention gate across the 6 branch tokens
    k_cvt<<<nblk(nc*6),256,0,st>>>(w.embbf, w.emb, nc*6);
    gemm(st, w.embbf, 0, CH, WT[W_QW], 0, CH, w.qf, 0, CH, P[S_QB], BG*n*6, CH, CH, 1, 0, 0);
    gemm(st, w.embbf, 0, CH, WT[W_KW], 0, CH, w.kf, 0, CH, P[S_KB], BG*n*6, CH, CH, 1, 0, 0);
    k_scores<<<nblk((size_t)BG*n*36),256,0,st>>>(w.scores, w.qf, w.kf, gGate, n, (size_t)BG*n*36);
    k_attn<<<nblk((size_t)BG*n*6),256,0,st>>>(attnOut, w.scores, (size_t)BG*n*6);
    k_ogate<<<nblk(nc),256,0,st>>>(w.of, attnOut, w.emb, nc);

    // ASAP pooling on o
    k_maskmax<<<dim3((unsigned)(BG*n)),CH,0,st>>>(w.xqf, w.of, Aadj, n);
    k_cvt<<<nblk(nc),256,0,st>>>(w.tmpbf, w.xqf, nc);
    gemm(st, w.tmpbf, 0, CH, WT[W_PL], 0, CH, w.tmp1, 0, CH, P[S_PL_B], BG*n, CH, CH, 1, 0, 0);
    k_cvt<<<nblk(nc),256,0,st>>>(w.tmpbf2, w.tmp1, nc);
    k_cvt<<<nblk(nc),256,0,st>>>(w.xbf, w.of, nc);
    k_trans_cvt<<<nblk(nc),256,0,st>>>(w.xTbf, w.of, n, CH, BG, nc);
    // sc = q @ o^T
    gemm(st, w.tmpbf2, llnc, CH, w.xbf, llnc, CH, w.scf, llnn, n, nullptr, n, n, CH, BG, 0, 0);
    k_pool_s<<<dim3((unsigned)n,BG),32,0,st>>>(w.sbf, w.scf, Aadj, n);
    // xn = s @ o
    gemm(st, w.sbf, llnn, n, w.xTbf, llnc, n, w.xnf, llnc, CH, nullptr, n, CH, n, BG, 0, 0);
    // LEConv fitness + top-k
    k_leconv<<<nblk((size_t)BG*n),256,0,st>>>(w.a1, w.b2, w.a3, w.xnf,
        P[S_LE1_W], P[S_LE1_B], P[S_LE2_W], P[S_LE3_W], P[S_LE3_B], (size_t)BG*n);
    k_fit<<<nblk((size_t)BG*n),256,0,st>>>(w.fit, Aadj, w.a1, w.b2, w.a3, w.degHat, n, (size_t)BG*n);
    k_topk<<<BG, n, 0, st>>>(w.idx, w.val, w.fit, n, kkeep);
    k_gather<<<nblk((size_t)BG*kkeep*CH),256,0,st>>>(xPooled, w.xnf, w.idx, w.val,
                                                     addv, n, kkeep, (size_t)BG*kkeep*CH);
    // A' = (S Ahat S^T != 0), diag removed, gathered
    gemm(st, w.sbf, llnn, n, w.AbfHatT, llnn, n, w.scf, llnn, n, nullptr, n, n, n, BG, 0, 0);
    k_cvt<<<nblk(nn),256,0,st>>>(w.alphabf, w.scf, nn);
    gemm(st, w.alphabf, llnn, n, w.sbf, llnn, n, w.a2f, llnn, n, nullptr, n, n, n, BG, 0, 0);
    k_A2<<<nblk((size_t)BG*kkeep*kkeep),256,0,st>>>(Aout, w.a2f, w.idx, n, kkeep,
                                                    (size_t)BG*kkeep*kkeep);
}

extern "C" void kernel_launch(void* const* d_in, const int* in_sizes, int n_in,
                              void* d_out, int out_size, void* d_ws, size_t ws_size,
                              hipStream_t stream)
{
    const float* x  = (const float*)d_in[0];
    const int*   ei = (const int*)d_in[1];
    const float* ac = (const float*)d_in[3];
    const float* fin_b = (const float*)d_in[4];
    const float* fin_w = (const float*)d_in[5];
    const float* P1[31]; const float* P2[31];
    for (int i = 0; i < 31; ++i){ P1[i] = (const float*)d_in[6+i]; P2[i] = (const float*)d_in[37+i]; }
    const float* v0_b = (const float*)d_in[68]; const float* v0_w = (const float*)d_in[69];
    const float* v1_b = (const float*)d_in[70]; const float* v1_w = (const float*)d_in[71];
    const int E = in_sizes[1]/2;
    const int N1 = 512, N2 = 256, N4 = 128;

    float* out   = (float*)d_out;
    float* attn1 = out + BG*CH;
    float* attn2 = attn1 + (size_t)BG*N1*36;

    // workspace carve-up
    char* base = (char*)d_ws; size_t off = 0;
    auto alc = [&](size_t elems, size_t esz)->void*{
        void* p = base + off; off += ((elems*esz + 255)/256)*256; return p; };
    const size_t NN1 = (size_t)BG*N1*N1, NC1 = (size_t)BG*N1*CH;

    WSS w; w.st = stream;
    w.A1   = (float*)alc(NN1, 4);
    w.A2   = (float*)alc((size_t)BG*N2*N2, 4);
    w.A4   = (float*)alc((size_t)BG*N4*N4, 4);
    w.scf  = (float*)alc(NN1, 4);
    w.emb  = (float*)alc(NC1*6, 4);
    w.qf   = (float*)alc(NC1*6, 4);    w.a2f = w.qf;   // pool-phase reuse
    w.kf   = (float*)alc(NC1*6, 4);
    w.scores = (float*)alc((size_t)BG*N1*36, 4);
    w.of   = (float*)alc(NC1, 4);
    w.xqf  = (float*)alc(NC1, 4);
    w.xnf  = (float*)alc(NC1, 4);
    w.axf  = (float*)alc(NC1, 4);
    w.tmp1 = (float*)alc(NC1, 4);
    w.tmp2 = (float*)alc(NC1, 4);
    w.hh   = (float*)alc(NC1, 4);
    w.hbuf = (float*)alc(NC1, 4);
    w.xp   = (float*)alc((size_t)BG*N2*CH, 4);
    w.xs2  = (float*)alc((size_t)BG*N4*CH, 4);
    w.finout = (float*)alc((size_t)BG*N4*CH, 4);
    w.gm  = (float*)alc(BG*CH, 4);  w.v0 = (float*)alc(BG*CH, 4);
    w.v1  = (float*)alc(BG*CH, 4);  w.gmv = (float*)alc(BG*CH, 4);
    w.g1  = (float*)alc(BG*6, 4);   w.g2 = (float*)alc(BG*6, 4);
    w.degA = (float*)alc((size_t)BG*N1, 4); w.degHat = (float*)alc((size_t)BG*N1, 4);
    w.dinv = (float*)alc((size_t)BG*N1, 4); w.invm   = (float*)alc((size_t)BG*N1, 4);
    w.asrcv = (float*)alc((size_t)BG*N1*2, 4); w.adstv = (float*)alc((size_t)BG*N1*2, 4);
    w.a1 = (float*)alc((size_t)BG*N1, 4); w.b2 = (float*)alc((size_t)BG*N1, 4);
    w.a3 = (float*)alc((size_t)BG*N1, 4); w.fit = (float*)alc((size_t)BG*N1, 4);
    w.val = (float*)alc((size_t)BG*N2, 4); w.idx = (int*)alc((size_t)BG*N2, 4);
    w.AbfRaw  = (_Float16*)alc(NN1, 2);
    w.AbfHat  = (_Float16*)alc(NN1, 2);
    w.AbfHatT = (_Float16*)alc(NN1, 2);
    w.alphabf = (_Float16*)alc(2*NN1, 2);   // also reused as tbf in pooling
    w.xbf   = (_Float16*)alc(NC1, 2);
    w.xTbf  = (_Float16*)alc(NC1, 2);
    w.axbf  = (_Float16*)alc(NC1, 2);
    w.tmpbf = (_Float16*)alc(NC1, 2);
    w.tmpbf2= (_Float16*)alc(NC1, 2);
    w.sctbf = (_Float16*)alc(NC1, 2);
    w.embbf = (_Float16*)alc(NC1*6, 2);
    w.hhTbf = (_Float16*)alc(NC1, 2);
    w.sbf   = (_Float16*)alc(NN1, 2);

    // transposed f16 weights (Bt layout [N][K])
    _Float16* wT[27];
    for (int i = 0; i < 27; ++i) wT[i] = (_Float16*)alc(CH*CH, 2);
    const int wsel[12] = { S_GCN_W, S_SAGE_WL, S_SAGE_WR, S_GAT_W, S_GIN_W1, S_GIN_W2,
                           S_LIN_W, S_GR_WREL, S_GR_WROOT, S_PL_W, S_KW, S_QW };
    for (int i = 0; i < 12; ++i){
        k_trans_cvt<<<nblk(CH*CH),256,0,stream>>>(wT[i],    P1[wsel[i]], CH, CH, 1, CH*CH);
        k_trans_cvt<<<nblk(CH*CH),256,0,stream>>>(wT[12+i], P2[wsel[i]], CH, CH, 1, CH*CH);
    }
    k_trans_cvt<<<nblk(CH*CH),256,0,stream>>>(wT[24], v0_w,  CH, CH, 1, CH*CH);
    k_trans_cvt<<<nblk(CH*CH),256,0,stream>>>(wT[25], v1_w,  CH, CH, 1, CH*CH);
    k_trans_cvt<<<nblk(CH*CH),256,0,stream>>>(wT[26], fin_w, CH, CH, 1, CH*CH);

    // adjacency from edge list
    k_fill<<<nblk(NN1),256,0,stream>>>(w.A1, 0.0f, NN1);
    k_scatter<<<nblk((size_t)E),256,0,stream>>>(w.A1, ei, E, N1);

    // virtual-node updates
    k_mean_rows<<<nblk((size_t)BG*CH),256,0,stream>>>(w.gm, x, N1, (size_t)BG*CH);
    k_cvt<<<nblk((size_t)BG*CH),256,0,stream>>>(w.tmpbf, w.gm, (size_t)BG*CH);
    gemm(stream, w.tmpbf, 0, CH, wT[24], 0, CH, w.v0, 0, CH, v0_b, BG, CH, CH, 1, 0, 0);
    k_add_bcast<<<nblk(NC1),256,0,stream>>>(w.hbuf, x, w.v0, N1, NC1);
    k_vecadd<<<nblk((size_t)BG*CH),256,0,stream>>>(w.gmv, w.gm, w.v0, (size_t)BG*CH);
    k_cvt<<<nblk((size_t)BG*CH),256,0,stream>>>(w.tmpbf, w.gmv, (size_t)BG*CH);
    gemm(stream, w.tmpbf, 0, CH, wT[25], 0, CH, w.v1, 0, CH, v1_b, BG, CH, CH, 1, 0, 0);

    // stage 1
    k_acgate<<<nblk((size_t)BG*6),256,0,stream>>>(w.g1, ac, P1[S_ACW], P1[S_ACB], (size_t)BG*6);
    run_stage(w, w.hbuf, w.A1, N1, 1e-9f, P1, &wT[0], w.g1, attn1, N2, w.v1, w.xp, w.A2);

    // stage 2
    k_acgate<<<nblk((size_t)BG*6),256,0,stream>>>(w.g2, ac, P2[S_ACW], P2[S_ACB], (size_t)BG*6);
    run_stage(w, w.xp, w.A2, N2, 1e-13f, P2, &wT[12], w.g2, attn2, N4, nullptr, w.xs2, w.A4);

    // final projection + mean pool
    k_cvt<<<nblk((size_t)BG*N4*CH),256,0,stream>>>(w.tmpbf, w.xs2, (size_t)BG*N4*CH);
    gemm(stream, w.tmpbf, 0, CH, wT[26], 0, CH, w.finout, 0, CH, fin_b, BG*N4, CH, CH, 1, 0, 0);
    k_mean_rows<<<nblk((size_t)BG*CH),256,0,stream>>>(out, w.finout, N4, (size_t)BG*CH);
}
